// GNNAnomalyDetector_55224689492027
// MI455X (gfx1250) — compile-verified, matrix operand order checked by reading, not run on hardware
//
#include <hip/hip_runtime.h>

// ---------------------------------------------------------------------------
// GNN edge-anomaly pipeline for MI455X (gfx1250, wave32, WMMA).
//
// Roofline: edge MLP = 39.4 GFLOP dominates (node GEMMs 3.3 GFLOP). HBM
// traffic ~200MB (~9us @ 23.3TB/s); node feature tables live in the 192MB L2
// so per-edge gathers are L2 hits -> compute-bound -> v_wmma_f32_16x16x32_f16.
// Node features are materialized once in packed f16 so the hot edge kernel
// gathers f16 directly (half the L2 bytes, no per-edge cvt VALU) and builds
// the z_src*z_dst term with v_pk_mul_f16. f32 is kept where semantics demand:
// segment-sum atomics, WMMA accumulators, LayerNorm statistics.
// ---------------------------------------------------------------------------

#define NN 100000
#define NE 1200000
#define H  64
#define EIN 32
#define CIN 224   // 3*H + EIN

typedef __attribute__((ext_vector_type(16))) _Float16 v16h;
typedef __attribute__((ext_vector_type(8)))  float    v8f;
typedef __attribute__((ext_vector_type(2)))  _Float16 h2v;

union Frag { v16h v; _Float16 h[16]; unsigned u[8]; uint4 q[2]; };

__device__ __forceinline__ unsigned pack2(float a, float b) {
  union { _Float16 h[2]; unsigned u; } t;
  t.h[0] = (_Float16)a; t.h[1] = (_Float16)b;
  return t.u;
}

__device__ __forceinline__ unsigned pkmul(unsigned a, unsigned b) {
  union { unsigned u; h2v h; } x, y, r;
  x.u = a; y.u = b; r.h = x.h * y.h;   // v_pk_mul_f16
  return r.u;
}

__device__ __forceinline__ v8f wmma_f16(const Frag& a, const Frag& b, v8f c) {
  return __builtin_amdgcn_wmma_f32_16x16x32_f16(false, a.v, false, b.v,
                                                (short)0, c, false, false);
}

// ---------------------------------------------------------------------------
__global__ void k_zero(float* __restrict__ p, int n) {
  int i = blockIdx.x * blockDim.x + threadIdx.x;
  if (i < n) p[i] = 0.0f;
}

__global__ void k_cvt16(const float* __restrict__ in, _Float16* __restrict__ o,
                        int n2) {               // n2 = element pairs
  int i = blockIdx.x * blockDim.x + threadIdx.x;
  if (i < n2) {
    float2 v = ((const float2*)in)[i];
    ((unsigned*)o)[i] = pack2(v.x, v.y);
  }
}

// Segment-sum scatter: 16 threads per edge, each handles 4 features (f32
// atomics to match reference accumulation semantics).
__global__ void k_scatter(const float* __restrict__ X,
                          const int* __restrict__ src,
                          const int* __restrict__ dst,
                          float* __restrict__ agg,
                          float* __restrict__ cnt,
                          int do_count) {
  int tid = blockIdx.x * blockDim.x + threadIdx.x;
  int e = tid >> 4;
  if (e >= NE) return;
  int f4 = (tid & 15) << 2;
  int s = src[e], d = dst[e];
  float4 v = *(const float4*)(X + (size_t)s * H + f4);
  float* ap = agg + (size_t)d * H + f4;
  atomicAdd(ap + 0, v.x);
  atomicAdd(ap + 1, v.y);
  atomicAdd(ap + 2, v.z);
  atomicAdd(ap + 3, v.w);
  if (do_count && f4 == 0) atomicAdd(cnt + d, 1.0f);
}

// ---------------------------------------------------------------------------
// SAGE layer: out = relu(LN(mean @ Wl + X @ Wr + b)). One wave per 16 nodes,
// 8 waves/block. X is consumed in packed f16; weights staged transposed in
// LDS as f16 so B fragments are two b128 LDS loads.
// Fragment layout (ISA 7.12.2, 16-bit A 16x32): lane (half = lane>>4) holds
// row lo = lane&15; VGPR p holds K pair (p>>2)*16 + half*8 + (p&3)*2, i.e.
// halfs [k0+half*8,+8) in u[0..3] and [k0+16+half*8,+8) in u[4..7].
__global__ void __launch_bounds__(256) k_sage(
    const _Float16* __restrict__ X16, const float* __restrict__ agg,
    const float* __restrict__ cnt,
    const float* __restrict__ wl, const float* __restrict__ wr,
    const float* __restrict__ bias,
    const float* __restrict__ gam, const float* __restrict__ bet,
    float* __restrict__ outf, _Float16* __restrict__ out16) {
  __shared__ __align__(16) _Float16 WtL[H][H];   // WtL[n][k] = wl[k][n]
  __shared__ __align__(16) _Float16 WtR[H][H];
  int t = threadIdx.x;
  for (int i = t; i < H * H; i += 256) {
    int n = i >> 6, k = i & 63;
    WtL[n][k] = (_Float16)wl[k * H + n];
    WtR[n][k] = (_Float16)wr[k * H + n];
  }
  __syncthreads();

  int w = t >> 5, lane = t & 31, lo = lane & 15, half = lane >> 4;
  int base = (blockIdx.x * 8 + w) * 16;
  int rowc = min(base + lo, NN - 1);          // clamp loads; stores predicated
  float inv = 1.0f / fmaxf(cnt[rowc], 1.0f);

  const v8f vz = {0, 0, 0, 0, 0, 0, 0, 0};
  v8f acc[4] = {vz, vz, vz, vz};

  const uint4*  xr = (const uint4*)(X16 + (size_t)rowc * H);
  const float4* ar = (const float4*)(agg + (size_t)rowc * H);

  for (int k0 = 0; k0 < H; k0 += 32) {
    Frag am, ax;
    ax.q[0] = xr[(k0 >> 3) + half];           // halfs [k0+half*8, +8)
    ax.q[1] = xr[(k0 >> 3) + 2 + half];       // halfs [k0+16+half*8, +8)
    {
      int i0 = (k0 >> 2) + (half << 1);
      float4 a0 = ar[i0],     a1 = ar[i0 + 1];
      float4 a2 = ar[i0 + 4], a3 = ar[i0 + 5];
      am.u[0] = pack2(a0.x * inv, a0.y * inv);
      am.u[1] = pack2(a0.z * inv, a0.w * inv);
      am.u[2] = pack2(a1.x * inv, a1.y * inv);
      am.u[3] = pack2(a1.z * inv, a1.w * inv);
      am.u[4] = pack2(a2.x * inv, a2.y * inv);
      am.u[5] = pack2(a2.z * inv, a2.w * inv);
      am.u[6] = pack2(a3.x * inv, a3.y * inv);
      am.u[7] = pack2(a3.z * inv, a3.w * inv);
    }
    for (int nt = 0; nt < 4; nt++) {
      int n = (nt << 4) + lo;
      Frag bl, br;
      bl.q[0] = *(const uint4*)&WtL[n][k0 + (half << 3)];
      bl.q[1] = *(const uint4*)&WtL[n][k0 + 16 + (half << 3)];
      br.q[0] = *(const uint4*)&WtR[n][k0 + (half << 3)];
      br.q[1] = *(const uint4*)&WtR[n][k0 + 16 + (half << 3)];
      acc[nt] = wmma_f16(am, bl, acc[nt]);
      acc[nt] = wmma_f16(ax, br, acc[nt]);
    }
  }

  // Bias + LayerNorm + ReLU. C/D layout: lane holds rows r+8*half, col lo+16nt.
  float bv[4], gv[4], ev[4];
  for (int nt = 0; nt < 4; nt++) {
    int n = (nt << 4) + lo;
    bv[nt] = bias[n]; gv[nt] = gam[n]; ev[nt] = bet[n];
  }
  for (int r = 0; r < 8; r++) {
    float vals[4], s = 0.f, sq = 0.f;
    for (int nt = 0; nt < 4; nt++) {
      float v = acc[nt][r] + bv[nt];
      vals[nt] = v; s += v; sq += v * v;
    }
    for (int m = 1; m < 16; m <<= 1) {        // reduce within 16-lane half
      s += __shfl_xor(s, m, 32);
      sq += __shfl_xor(sq, m, 32);
    }
    float mu = s * (1.0f / 64.0f);
    float var = sq * (1.0f / 64.0f) - mu * mu;
    float rs = rsqrtf(var + 1e-5f);
    int node = base + r + (half << 3);
    if (node < NN) {
      for (int nt = 0; nt < 4; nt++) {
        float o = fmaxf((vals[nt] - mu) * rs * gv[nt] + ev[nt], 0.0f);
        int col = (nt << 4) + lo;
        if (outf) outf[(size_t)node * H + col] = o;
        out16[(size_t)node * H + col] = (_Float16)o;
      }
    }
  }
}

// ---------------------------------------------------------------------------
// Edge MLP: logits = (relu(relu([zs|zd|zs*zd|ea] @ W1 + b1) @ W2 + b2)) @ W3 + b3
// One wave per 16 edges, 4 waves/block. 28 + 4 WMMA per wave. z gathered as
// packed f16 (L2-resident): staging is straight b128 copies + v_pk_mul_f16.
__global__ void __launch_bounds__(128) k_edge(
    const _Float16* __restrict__ Z16,
    const int* __restrict__ src, const int* __restrict__ dst,
    const float* __restrict__ EA,
    const float* __restrict__ cw1, const float* __restrict__ cb1,
    const float* __restrict__ cw2, const float* __restrict__ cb2,
    const float* __restrict__ cw3, const float* __restrict__ cb3,
    float* __restrict__ out) {
  __shared__ __align__(16) _Float16 Wt1[H][CIN];       // 28672 B (W1^T)
  __shared__ __align__(16) _Float16 Wt2[32][H];        //  4096 B (W2^T)
  __shared__ __align__(16) _Float16 feat[4][16][CIN];  // 28672 B per-wave stage

  int t = threadIdx.x;
  for (int i = t; i < H * CIN; i += 128) {
    int n = i / CIN, k = i % CIN;
    Wt1[n][k] = (_Float16)cw1[k * H + n];
  }
  for (int i = t; i < 32 * H; i += 128) {
    int n = i >> 6, k = i & 63;
    Wt2[n][k] = (_Float16)cw2[k * 32 + n];
  }
  __syncthreads();

  int w = t >> 5, lane = t & 31, lo = lane & 15, half = lane >> 4;
  int eBase = (blockIdx.x * 4 + w) * 16;

  // Stage edge representation (f16): [zs | zd | zs*zd | ea]. Two lanes per
  // edge; each handles one 32-feature half. zs/zd are packed-f16 row copies.
  {
    int e = eBase + lo;
    int s = src[e], d = dst[e];
    const uint4* zsp = (const uint4*)(Z16 + (size_t)s * H) + (half << 2);
    const uint4* zdp = (const uint4*)(Z16 + (size_t)d * H) + (half << 2);
    uint4* rowv = (uint4*)&feat[w][lo][0];     // uint4 = 8 f16 features
    for (int i = 0; i < 4; i++) {
      uint4 a = zsp[i];
      uint4 b = zdp[i];
      rowv[(half << 2) + i]      = a;          // zs  at f = half*32
      rowv[8 + (half << 2) + i]  = b;          // zd  at 64 + f
      uint4 p;
      p.x = pkmul(a.x, b.x); p.y = pkmul(a.y, b.y);
      p.z = pkmul(a.z, b.z); p.w = pkmul(a.w, b.w);
      rowv[16 + (half << 2) + i] = p;          // zs*zd at 128 + f
    }
    const float4* eap = (const float4*)(EA + (size_t)e * EIN + (half << 4));
    for (int i = 0; i < 2; i++) {
      float4 a = eap[2 * i], b = eap[2 * i + 1];
      uint4 q;
      q.x = pack2(a.x, a.y); q.y = pack2(a.z, a.w);
      q.z = pack2(b.x, b.y); q.w = pack2(b.z, b.w);
      rowv[24 + (half << 1) + i] = q;          // ea at 192 + half*16
    }
  }
  __syncthreads();

  const v8f vz = {0, 0, 0, 0, 0, 0, 0, 0};

  // GEMM1: [16 x 224] @ [224 x 64] -> 7 K-chunks x 4 N-tiles = 28 WMMA.
  v8f acc[4] = {vz, vz, vz, vz};
  for (int ch = 0; ch < 7; ch++) {
    int k0 = ch * 32;
    Frag a;
    a.q[0] = *(const uint4*)&feat[w][lo][k0 + (half << 3)];
    a.q[1] = *(const uint4*)&feat[w][lo][k0 + 16 + (half << 3)];
    for (int nt = 0; nt < 4; nt++) {
      int n = (nt << 4) + lo;
      Frag b;
      b.q[0] = *(const uint4*)&Wt1[n][k0 + (half << 3)];
      b.q[1] = *(const uint4*)&Wt1[n][k0 + 16 + (half << 3)];
      acc[nt] = wmma_f16(a, b, acc[nt]);
    }
  }

  // Bias + ReLU -> h1, restaged (D-layout -> row-major f16) over the
  // already-consumed head of this wave's feat region (same-wave LDS in-order).
  float cb1v[4];
  for (int nt = 0; nt < 4; nt++) cb1v[nt] = cb1[(nt << 4) + lo];
  for (int nt = 0; nt < 4; nt++)
    for (int r = 0; r < 8; r++) {
      float v = fmaxf(acc[nt][r] + cb1v[nt], 0.0f);
      feat[w][r + (half << 3)][(nt << 4) + lo] = (_Float16)v;
    }

  // GEMM2: [16 x 64] @ [64 x 32] -> 2 K-chunks x 2 N-tiles = 4 WMMA.
  v8f acc2[2] = {vz, vz};
  for (int k0 = 0; k0 < H; k0 += 32) {
    Frag a;
    a.q[0] = *(const uint4*)&feat[w][lo][k0 + (half << 3)];
    a.q[1] = *(const uint4*)&feat[w][lo][k0 + 16 + (half << 3)];
    for (int nt = 0; nt < 2; nt++) {
      int n = (nt << 4) + lo;
      Frag b;
      b.q[0] = *(const uint4*)&Wt2[n][k0 + (half << 3)];
      b.q[1] = *(const uint4*)&Wt2[n][k0 + 16 + (half << 3)];
      acc2[nt] = wmma_f16(a, b, acc2[nt]);
    }
  }

  // Final 32->1 dot + bias, reduced across the 16-lane half.
  float cb2v[2], w3v[2];
  for (int nt = 0; nt < 2; nt++) {
    cb2v[nt] = cb2[(nt << 4) + lo];
    w3v[nt] = cw3[(nt << 4) + lo];
  }
  float b3 = cb3[0];
  float partial[8];
  for (int r = 0; r < 8; r++) {
    float s = 0.f;
    for (int nt = 0; nt < 2; nt++)
      s += fmaxf(acc2[nt][r] + cb2v[nt], 0.0f) * w3v[nt];
    for (int m = 1; m < 16; m <<= 1) s += __shfl_xor(s, m, 32);
    partial[r] = s;
  }
  if (lo == 0)
    for (int r = 0; r < 8; r++)
      out[eBase + r + (half << 3)] = partial[r] + b3;
}

// ---------------------------------------------------------------------------
extern "C" void kernel_launch(void* const* d_in, const int* in_sizes, int n_in,
                              void* d_out, int out_size, void* d_ws,
                              size_t ws_size, hipStream_t stream) {
  (void)in_sizes; (void)n_in; (void)out_size; (void)ws_size;
  const float* x   = (const float*)d_in[0];
  const int*   ei  = (const int*)d_in[1];   // [2][NE] (int32 per harness rule)
  const float* ea  = (const float*)d_in[2];
  const float* wl0 = (const float*)d_in[3];
  const float* wr0 = (const float*)d_in[4];
  const float* b0  = (const float*)d_in[5];
  const float* g0  = (const float*)d_in[6];
  const float* e0  = (const float*)d_in[7];
  const float* wl1 = (const float*)d_in[8];
  const float* wr1 = (const float*)d_in[9];
  const float* b1  = (const float*)d_in[10];
  const float* g1  = (const float*)d_in[11];
  const float* e1  = (const float*)d_in[12];
  const float* cw1 = (const float*)d_in[13];
  const float* cb1 = (const float*)d_in[14];
  const float* cw2 = (const float*)d_in[15];
  const float* cb2 = (const float*)d_in[16];
  const float* cw3 = (const float*)d_in[17];
  const float* cb3 = (const float*)d_in[18];
  const int* src = ei;
  const int* dst = ei + NE;

  // Workspace carve-out (~77.2 MB): agg(f32) | cnt(f32) | h(f32) | x16/z16 | h16
  // x16 is dead after sage0 -> reused as z16.
  float*    agg = (float*)d_ws;
  float*    cnt = agg + (size_t)NN * H;
  float*    hbf = cnt + NN;
  _Float16* x16 = (_Float16*)(hbf + (size_t)NN * H);
  _Float16* h16 = x16 + (size_t)NN * H;
  _Float16* z16 = x16;                       // alias (x16 dead after sage0)

  int nz = NN * (H + 1);
  k_zero<<<(nz + 255) / 256, 256, 0, stream>>>(agg, nz);
  k_cvt16<<<(NN * H / 2 + 255) / 256, 256, 0, stream>>>(x, x16, NN * H / 2);
  k_scatter<<<(NE * 16 + 255) / 256, 256, 0, stream>>>(x, src, dst, agg, cnt, 1);
  k_sage<<<(NN + 127) / 128, 256, 0, stream>>>(x16, agg, cnt, wl0, wr0, b0,
                                               g0, e0, hbf, h16);
  k_zero<<<(NN * H + 255) / 256, 256, 0, stream>>>(agg, NN * H);
  k_scatter<<<(NE * 16 + 255) / 256, 256, 0, stream>>>(hbf, src, dst, agg, cnt, 0);
  k_sage<<<(NN + 127) / 128, 256, 0, stream>>>(h16, agg, cnt, wl1, wr1, b1,
                                               g1, e1, nullptr, z16);
  k_edge<<<NE / 64, 128, 0, stream>>>(z16, src, dst, ea, cw1, cb1, cw2, cb2,
                                      cw3, cb3, (float*)d_out);
}
